// TiledEquivariantMLP_68745246540042
// MI455X (gfx1250) — matrix operand order repaired and checked by reading, not compiled, not run on hardware
//
#include <hip/hip_runtime.h>

typedef __attribute__((ext_vector_type(16))) _Float16 v16h;
typedef __attribute__((ext_vector_type(8)))  float    v8f;

namespace {

constexpr int T_CLS = 10;
constexpr int BATCH = 32;
constexpr int NATOM = 512;
constexpr int OUT_D = 128;
constexpr int HID_D = 64;
constexpr int D_IN  = 272;
constexpr int TOUT  = T_CLS * OUT_D;

// 16-bit A-matrix 16x32 per-lane K mapping (CDNA5 ISA §7.12.2):
// lanes 0-15 hold K {0..7, 16..23}, lanes 16-31 hold K {8..15, 24..31}.
__device__ __forceinline__ int a_k(int e, int hi) {
  return e + hi * 8 + (e >= 8 ? 8 : 0);
}

// Load an A fragment (16 rows x 32 K) from a row-major f16 LDS tile.
__device__ __forceinline__ v16h lds_A(const _Float16* buf, int stride, int k0, int lane) {
  const int row = lane & 15;
  const int hi  = lane >> 4;
  v16h a;
#pragma unroll
  for (int e = 0; e < 16; ++e) a[e] = buf[row * stride + k0 + a_k(e, hi)];
  return a;
}

// Load a B fragment (32 K x 16 N) from a row-major f32 global weight matrix,
// folding the 1/sqrt(fan_in) e3nn norm into the weights (linear, so legal).
// lanes 0-15: col = lane, K = k0+{0..7,16..23} ; lanes 16-31: col = lane-16, K = k0+{8..15,24..31}.
__device__ __forceinline__ v16h glb_B(const float* __restrict__ W, int ldw,
                                      int k0, int n0, int lane, float scale) {
  const int col = lane & 15;
  const int kb  = k0 + ((lane >> 4) << 3);
  v16h b;
#pragma unroll
  for (int e = 0; e < 16; ++e)
    b[e] = (_Float16)(W[(kb + e + (e >= 8 ? 8 : 0)) * ldw + n0 + col] * scale);
  return b;
}

__device__ __forceinline__ v8f wmma_f16(v16h a, v16h b, v8f c) {
  return __builtin_amdgcn_wmma_f32_16x16x32_f16(false, a, false, b, (short)0, c,
                                                false, false);
}

// Fast SiLU: v_exp_f32 + v_rcp_f32 (TRANS ops co-issue with WMMA), no IEEE
// division Newton iteration.
__device__ __forceinline__ float silu(float v) {
  const float e = __expf(-v);
  return v * __builtin_amdgcn_rcpf(1.0f + e);
}

} // namespace

extern "C" __global__ __launch_bounds__(128)
void equiv_mlp_kernel(const float* __restrict__ x,      // (B, NA, 272)
                      const float* __restrict__ W0,     // (T, 32, 256) - cols 0..63 live
                      const float* __restrict__ W1,     // (T, 64, 256) - cols 0..63 live
                      const float* __restrict__ Wo,     // (T, 64, 128)
                      const int*   __restrict__ mapping,// (B, NA, 1)
                      float* __restrict__ out) {        // (B, T*128)
  const int t    = blockIdx.x;
  const int b    = blockIdx.y;
  const int wave = threadIdx.x >> 5;
  const int lane = threadIdx.x & 31;
  const int hi   = lane >> 4;
  const int col  = lane & 15;

  __shared__ _Float16 xs[4][16 * 32];
  __shared__ _Float16 h1[4][16 * HID_D];
  __shared__ _Float16 h2[4][16 * HID_D];
  __shared__ int      mapb[4][16];
  __shared__ float    oacc[4][OUT_D];

  const float* W0t = W0 + (size_t)t * 32 * 256;
  const float* W1t = W1 + (size_t)t * 64 * 256;
  const float* Wot = Wo + (size_t)t * 64 * 128;

  const float n0 = 0.17677669529663687f; // 1/sqrt(32), folded into B0
  const float n1 = 0.125f;               // 1/sqrt(64), folded into B1 / mask

  // Block-invariant weight B-fragments, hoisted into VGPRs (norms pre-folded).
  v16h B0[4], B1[8], B2[16];
#pragma unroll
  for (int nt = 0; nt < 4; ++nt) B0[nt] = glb_B(W0t, 256, 0, nt * 16, lane, n0);
#pragma unroll
  for (int nt = 0; nt < 4; ++nt) {
    B1[nt * 2 + 0] = glb_B(W1t, 256, 0,  nt * 16, lane, n1);
    B1[nt * 2 + 1] = glb_B(W1t, 256, 32, nt * 16, lane, n1);
  }
#pragma unroll
  for (int nt = 0; nt < 8; ++nt) {
    B2[nt * 2 + 0] = glb_B(Wot, 128, 0,  nt * 16, lane, 1.0f);
    B2[nt * 2 + 1] = glb_B(Wot, 128, 32, nt * 16, lane, 1.0f);
  }

  float acc[8];
#pragma unroll
  for (int i = 0; i < 8; ++i) acc[i] = 0.0f;

  for (int tile = wave; tile < NATOM / 16; tile += 4) {
    const int atom0 = tile * 16;

    // Stage the live 16x32 scalar block of this atom tile into LDS as f16
    // (coalesced: lanes sweep consecutive k within a row).
#pragma unroll
    for (int i = 0; i < 16; ++i) {
      const int idx = i * 32 + lane;
      const int ar  = idx >> 5;
      const int k   = idx & 31;
      xs[wave][ar * 32 + k] =
          (_Float16)x[((size_t)(b * NATOM + atom0 + ar)) * D_IN + k];
    }
    if (lane < 16) mapb[wave][lane] = mapping[b * NATOM + atom0 + lane];
    asm volatile("s_wait_dscnt 0" ::: "memory");

    v16h A0 = lds_A(xs[wave], 32, 0, lane);

    // Layer 0 (live scalar path): h1 = silu(x_s @ (W0[:, :64]/sqrt32))
#pragma unroll
    for (int nt = 0; nt < 4; ++nt) {
      v8f c = {};
      c = wmma_f16(A0, B0[nt], c);
#pragma unroll
      for (int r = 0; r < 8; ++r) {
        const int row = r + hi * 8;              // f32 C/D layout, §7.12.2
        h1[wave][row * HID_D + nt * 16 + col] = (_Float16)silu(c[r]);
      }
    }
    asm volatile("s_wait_dscnt 0" ::: "memory");

    v16h A1a = lds_A(h1[wave], HID_D, 0, lane);
    v16h A1b = lds_A(h1[wave], HID_D, 32, lane);

    // Layer 1: h2 = silu(h1 @ (W1[:, :64]/8))
#pragma unroll
    for (int nt = 0; nt < 4; ++nt) {
      v8f c = {};
      c = wmma_f16(A1a, B1[nt * 2 + 0], c);
      c = wmma_f16(A1b, B1[nt * 2 + 1], c);
#pragma unroll
      for (int r = 0; r < 8; ++r) {
        const int row = r + hi * 8;
        h2[wave][row * HID_D + nt * 16 + col] = (_Float16)silu(c[r]);
      }
    }
    asm volatile("s_wait_dscnt 0" ::: "memory");

    v16h A2a = lds_A(h2[wave], HID_D, 0, lane);
    v16h A2b = lds_A(h2[wave], HID_D, 32, lane);

    // Per-row class mask, with the 1/8 output norm folded in (linear).
    float msk[8];
#pragma unroll
    for (int r = 0; r < 8; ++r)
      msk[r] = (mapb[wave][r + hi * 8] == t) ? n1 : 0.0f;

    // Output layer + masked reduction over the tile's 16 atoms.
#pragma unroll
    for (int nt = 0; nt < 8; ++nt) {
      v8f c = {};
      c = wmma_f16(A2a, B2[nt * 2 + 0], c);
      c = wmma_f16(A2b, B2[nt * 2 + 1], c);
      float p = 0.0f;
#pragma unroll
      for (int r = 0; r < 8; ++r) p += c[r] * msk[r];
      acc[nt] += p;
    }
  }

  // Lanes L and L+16 hold the same output column (rows 0-7 vs 8-15): fold.
#pragma unroll
  for (int nt = 0; nt < 8; ++nt) acc[nt] += __shfl_xor(acc[nt], 16, 32);
  if (lane < 16) {
#pragma unroll
    for (int nt = 0; nt < 8; ++nt) oacc[wave][nt * 16 + col] = acc[nt];
  }
  __syncthreads();

  // Block exclusively owns out[b, t*128 : (t+1)*128] -> deterministic store.
  for (int i = threadIdx.x; i < OUT_D; i += 128) {
    const float s = oacc[0][i] + oacc[1][i] + oacc[2][i] + oacc[3][i];
    out[(size_t)b * TOUT + t * OUT_D + i] = s;
  }
}

extern "C" void kernel_launch(void* const* d_in, const int* in_sizes, int n_in,
                              void* d_out, int out_size, void* d_ws, size_t ws_size,
                              hipStream_t stream) {
  (void)in_sizes; (void)n_in; (void)out_size; (void)d_ws; (void)ws_size;
  const float* x   = (const float*)d_in[0];   // atom_features
  const float* W0  = (const float*)d_in[1];   // W0_0
  const float* W1  = (const float*)d_in[5];   // W1_0
  const float* Wo  = (const float*)d_in[9];   // Wout
  const int*   map = (const int*)d_in[10];    // mlp_mapping
  float* out = (float*)d_out;

  dim3 grid(T_CLS, BATCH);
  equiv_mlp_kernel<<<grid, 128, 0, stream>>>(x, W0, W1, Wo, map, out);
}
